// RelativePositionalEmbedding_37349035606582
// MI455X (gfx1250) — matrix operand order repaired and checked by reading, not compiled, not run on hardware
//
#include <hip/hip_runtime.h>

#define DI __device__ __forceinline__

typedef unsigned short u16;
typedef unsigned int   u32;

typedef __attribute__((ext_vector_type(16))) __bf16 v16bf;
typedef __attribute__((ext_vector_type(8)))  float  v8f;
typedef __attribute__((ext_vector_type(8)))  u32    v8u;
typedef __attribute__((ext_vector_type(4)))  u32    v4u;

namespace {
constexpr int kB   = 2;
constexpr int kS   = 2048;
constexpr int kE   = 1024;
constexpr int kH   = 16;
constexpr int kD   = 64;
constexpr int kP   = 128;
constexpr int kR   = 2 * kP + 1;   // 257 relative buckets
constexpr int kRp  = 288;          // bucket K padded to 9*32 for WMMA
constexpr int kQp  = 272;          // qdot N padded to 17*16
constexpr int kM   = kB * kS;      // 4096 rows for the big GEMMs
}

DI u16 f2bf(float f) {
  u32 u = __float_as_uint(f);
  return (u16)((u + 0x7FFFu + ((u >> 16) & 1u)) >> 16);   // RNE f32->bf16
}

DI v16bf asbf(v8u x) { return __builtin_bit_cast(v16bf, x); }

DI v8f wmma_bf16(v16bf a, v16bf b, v8f c) {
  return __builtin_amdgcn_wmma_f32_16x16x32_bf16(false, a, false, b, (short)0, c,
                                                 false, false);
}

// A fragment (16x32 bf16), row-major bf16 source. ISA layout: lane m = lane&15,
// K-half hk = (lane>>4)*8; VGPRs 0-3 hold K hk..hk+7, VGPRs 4-7 hold K 16+hk..23+hk.
// Both halves are contiguous 16-byte blocks -> two B128 loads.
DI v8u ldA_bf(const u16* base, int stride, int lane) {
  const u16* row = base + (size_t)(lane & 15) * stride + (lane >> 4) * 8;
  v4u a = *(const v4u*)(row);
  v4u b = *(const v4u*)(row + 16);
  v8u r;
  r[0] = a[0]; r[1] = a[1]; r[2] = a[2]; r[3] = a[3];
  r[4] = b[0]; r[5] = b[1]; r[6] = b[2]; r[7] = b[3];
  return r;
}

// A fragment from row-major f32 with on-the-fly bf16 conversion (small uses only).
DI v8u ldA_f32(const float* base, int stride, int lane) {
  const float* row = base + (size_t)(lane & 15) * stride;
  const int hk = (lane >> 4) * 8;
  v8u r;
#pragma unroll
  for (int i = 0; i < 8; ++i) {
    int k0 = ((i < 4) ? 2 * i : 8 + 2 * i) + hk;
    u32 lo = f2bf(row[k0]);
    u32 hi = f2bf(row[k0 + 1]);
    r[i] = lo | (hi << 16);
  }
  return r;
}

// B fragment (32x16 bf16): colPtr = this lane's column, contiguous along K.
// Same K layout as A: hk = (lane>>4)*8; VGPRs 0-3 K hk..hk+7, VGPRs 4-7 K 16+hk..23+hk.
DI v8u ldB_bf(const u16* colPtr, int lane) {
  const u16* p = colPtr + (lane >> 4) * 8;
  v4u a = *(const v4u*)(p);
  v4u b = *(const v4u*)(p + 16);
  v8u r;
  r[0] = a[0]; r[1] = a[1]; r[2] = a[2]; r[3] = a[3];
  r[4] = b[0]; r[5] = b[1]; r[6] = b[2]; r[7] = b[3];
  return r;
}

// ---------------- conversion kernels ----------------

__global__ void __launch_bounds__(256)
cvt_bf16_kernel(const float* __restrict__ src, u16* __restrict__ dst, int n) {
  int i = blockIdx.x * 256 + threadIdx.x;
  if (i < n) dst[i] = f2bf(src[i]);
}

// rv_table [257][64] -> transposed + K-padded [64][288] bf16
__global__ void __launch_bounds__(256)
cvt_rvT_kernel(const float* __restrict__ rv, u16* __restrict__ dst) {
  int i = blockIdx.x * 256 + threadIdx.x;
  if (i < kD * kRp) {
    int dd = i / kRp, r = i % kRp;
    dst[i] = (r < kR) ? f2bf(rv[r * kD + dd]) : (u16)0;
  }
}

// ---------------- QKV projection GEMM ----------------
// Y = X @ W^T + b ; X bf16 [4096,1024] (pre-converted), W bf16 [1024,1024].
// Block tile 64x256, 8 waves as 2(m) x 4(n), wave tile 32x64 (acc[2][4]):
// 12 B128 loads per 8 WMMAs. z = 0/1/2 selects q/k/v. qh,kh scattered to
// [B,H,S,d]; v stored transposed [B,H,d,S] for K-contiguous PV B-fragments.
__global__ void __launch_bounds__(256)
proj_kernel(const u16* __restrict__ xq, const u16* __restrict__ xk,
            const u16* __restrict__ xv, const u16* __restrict__ wq,
            const u16* __restrict__ wk, const u16* __restrict__ wv,
            const float* __restrict__ bq, const float* __restrict__ bk,
            const float* __restrict__ bv, u16* __restrict__ qh,
            u16* __restrict__ kh, u16* __restrict__ vhT) {
  const int lane = threadIdx.x & 31;
  const int wave = threadIdx.x >> 5;
  const int z = blockIdx.z;
  const u16*   X    = (z == 0) ? xq : (z == 1) ? xk : xv;
  const u16*   W    = (z == 0) ? wq : (z == 1) ? wk : wv;
  const float* bias = (z == 0) ? bq : (z == 1) ? bk : bv;

  const int mBase = blockIdx.x * 64 + (wave >> 2) * 32;
  const int nBase = blockIdx.y * 256 + (wave & 3) * 64;
  const int nl = lane & 15, hi16 = lane >> 4;

  v8f acc[2][4] = {};
  for (int k0 = 0; k0 < kE; k0 += 32) {
    v16bf a0 = asbf(ldA_bf(X + (size_t)mBase * kE + k0, kE, lane));
    v16bf a1 = asbf(ldA_bf(X + (size_t)(mBase + 16) * kE + k0, kE, lane));
    v16bf b[4];
#pragma unroll
    for (int ni = 0; ni < 4; ++ni)
      b[ni] = asbf(ldB_bf(W + (size_t)(nBase + ni * 16 + nl) * kE + k0, lane));
#pragma unroll
    for (int ni = 0; ni < 4; ++ni) {
      acc[0][ni] = wmma_bf16(a0, b[ni], acc[0][ni]);
      acc[1][ni] = wmma_bf16(a1, b[ni], acc[1][ni]);
    }
  }
#pragma unroll
  for (int mi = 0; mi < 2; ++mi)
#pragma unroll
    for (int ni = 0; ni < 4; ++ni)
#pragma unroll
      for (int j = 0; j < 8; ++j) {
        int m = mBase + mi * 16 + j + 8 * hi16;
        int n = nBase + ni * 16 + nl;
        u16 o = f2bf(acc[mi][ni][j] + bias[n]);
        int bb = m >> 11, ss = m & (kS - 1);
        int hh = n >> 6, dd = n & (kD - 1);
        if (z == 2)
          vhT[((size_t)(bb * kH + hh) * kD + dd) * kS + ss] = o;
        else {
          u16* dst = (z == 0) ? qh : kh;
          dst[((size_t)(bb * kH + hh) * kS + ss) * kD + dd] = o;
        }
      }
}

// ---------------- fused relative attention ----------------
// One workgroup (8 wave32) per (b, h, 16-query tile). Full 16x2048 score strip
// in LDS; waves split K into 256-wide slices. Probs kept unnormalized; the
// 1/rowsum is applied once in the combine.
__global__ void __launch_bounds__(256)
attn_kernel(const u16* __restrict__ qh, const u16* __restrict__ kh,
            const u16* __restrict__ vhT, const u16* __restrict__ rkB,
            const u16* __restrict__ rvT, u16* __restrict__ ctx) {
  extern __shared__ char smem[];
  float* scores  = (float*)smem;              // 16*2048 f32   (131072 B)
  u16*   pbf     = (u16*)(scores + 16 * kS);  // 16*2048 bf16  ( 65536 B)
  u16*   qtile   = pbf + 16 * kS;             // 16*64   bf16  (  2048 B)
  float* qdot    = (float*)(qtile + 16 * kD); // 16*272  f32   ( 17408 B)
  float* buckets = qdot + 16 * kQp;           // 16*288  f32   ( 18432 B)
  float* wsum    = buckets + 16 * kRp;        // 8*16*64 f32   ( 32768 B)
  float* w2buf   = wsum + 8 * 16 * 64;        // 16*64   f32   (  4096 B)
  float* rowinv  = w2buf + 16 * kD;           // 16      f32   (    64 B)

  const int t = threadIdx.x;
  const int lane = t & 31, wave = t >> 5;
  const int nl = lane & 15, hi16 = lane >> 4;
  const int bh = blockIdx.x >> 7;             // 128 q-tiles per (b,h)
  const int q0 = (blockIdx.x & 127) * 16;
  const size_t bhS = (size_t)bh * kS;

  {  // stage Q tile; zero bucket accumulators (incl. K-pad region)
    const u16* src = qh + (bhS + q0) * kD;
    for (int i = t; i < 16 * kD; i += 256) qtile[i] = src[i];
    for (int i = t; i < 16 * kRp; i += 256) buckets[i] = 0.f;
  }
  __syncthreads();

  v16bf aq0 = asbf(ldA_bf(qtile, kD, lane));       // Q K-dims 0..31
  v16bf aq1 = asbf(ldA_bf(qtile + 32, kD, lane));  // Q K-dims 32..63

  // qdot[16 x 272] = Qtile @ rk_table^T  (17 N-tiles over 8 waves)
  for (int nt = wave; nt < 17; nt += 8) {
    int r = nt * 16 + nl;
    v8u p0 = {}, p1 = {};
    if (r < kR) {
      const u16* colPtr = rkB + (size_t)r * kD;
      p0 = ldB_bf(colPtr, lane);
      p1 = ldB_bf(colPtr + 32, lane);
    }
    v8f acc = {};
    acc = wmma_bf16(aq0, asbf(p0), acc);
    acc = wmma_bf16(aq1, asbf(p1), acc);
#pragma unroll
    for (int j = 0; j < 8; ++j)
      qdot[(j + 8 * hi16) * kQp + nt * 16 + nl] = acc[j];
  }
  __syncthreads();

  // scores = (Q K^T + qdot-gather) / sqrt(d); each wave covers 16 k-tiles
  for (int kt = wave * 16; kt < wave * 16 + 16; ++kt) {
    int kbase = kt * 16;
    const u16* colPtr = kh + (bhS + kbase + nl) * kD;
    v8f acc = {};
    acc = wmma_bf16(aq0, asbf(ldB_bf(colPtr, lane)), acc);
    acc = wmma_bf16(aq1, asbf(ldB_bf(colPtr + 32, lane)), acc);
#pragma unroll
    for (int j = 0; j < 8; ++j) {
      int m = j + 8 * hi16;
      int kg = kbase + nl;
      int rel = kg - (q0 + m);
      rel = (rel < -kP) ? -kP : (rel > kP) ? kP : rel;
      scores[m * kS + kg] = (acc[j] + qdot[m * kQp + rel + kP]) * 0.125f;
    }
  }
  __syncthreads();

  // softmax: 2 rows per wave. One sweep for max; one fused sweep doing
  // exp + bf16-prob store + bucket atomics + row-sum. No normalize sweep.
#pragma unroll
  for (int rr = 0; rr < 2; ++rr) {
    int row = wave * 2 + rr;
    float* srow = scores + row * kS;
    u16*   prow = pbf + row * kS;
    float mx = -3.0e38f;
    for (int c = lane; c < kS; c += 32) mx = fmaxf(mx, srow[c]);
#pragma unroll
    for (int off = 16; off > 0; off >>= 1) mx = fmaxf(mx, __shfl_xor(mx, off, 32));
    float sum = 0.f;
    for (int c = lane; c < kS; c += 32) {
      float e = __expf(srow[c] - mx);
      prow[c] = f2bf(e);
      sum += e;
      int rel = c - (q0 + row);
      rel = (rel < -kP) ? -kP : (rel > kP) ? kP : rel;
      atomicAdd(&buckets[row * kRp + rel + kP], e);
    }
#pragma unroll
    for (int off = 16; off > 0; off >>= 1) sum += __shfl_xor(sum, off, 32);
    if (lane == 0) rowinv[row] = 1.f / sum;
  }
  __syncthreads();

  // w1 partials (unnormalized): P(16 x 256-slice) @ V(256 x 64), per wave
  {
    v8f acc[4] = {};
    for (int ks = 0; ks < 8; ++ks) {
      int kbase = wave * 256 + ks * 32;
      v16bf a = asbf(ldA_bf(pbf + kbase, kS, lane));
#pragma unroll
      for (int ni = 0; ni < 4; ++ni) {
        const u16* colPtr = vhT + ((size_t)bh * kD + ni * 16 + nl) * kS + kbase;
        acc[ni] = wmma_bf16(a, asbf(ldB_bf(colPtr, lane)), acc[ni]);
      }
    }
#pragma unroll
    for (int ni = 0; ni < 4; ++ni)
#pragma unroll
      for (int j = 0; j < 8; ++j)
        wsum[wave * 1024 + (j + 8 * hi16) * kD + ni * 16 + nl] = acc[ni][j];
  }
  __syncthreads();

  // w2 (unnormalized) = buckets(16x288) @ rv(288x64): waves 0..3, one 16-col tile each
  if (wave < 4) {
    v8f acc = {};
    for (int ks = 0; ks < 9; ++ks) {
      int kbase = ks * 32;
      v16bf a = asbf(ldA_f32(buckets + kbase, kRp, lane));
      const u16* colPtr = rvT + (size_t)(wave * 16 + nl) * kRp + kbase;
      acc = wmma_bf16(a, asbf(ldB_bf(colPtr, lane)), acc);
    }
#pragma unroll
    for (int j = 0; j < 8; ++j)
      w2buf[(j + 8 * hi16) * kD + wave * 16 + nl] = acc[j];
  }
  __syncthreads();

  // ctx = (w1 + w2) * rowinv, stored bf16 [B,S,E] row-major
  {
    int b = bh >> 4, h = bh & 15;
    for (int idx = t; idx < 16 * kD; idx += 256) {
      int m = idx >> 6, dd = idx & (kD - 1);
      float val = w2buf[idx];
#pragma unroll
      for (int w = 0; w < 8; ++w) val += wsum[w * 1024 + idx];
      val *= rowinv[m];
      ctx[(size_t)(b * kS + q0 + m) * kE + h * kD + dd] = f2bf(val);
    }
  }
}

// ---------------- output projection GEMM ----------------
// Same 64x256 block / 32x64 wave tiling as proj_kernel; fp32 output + bias.
__global__ void __launch_bounds__(256)
out_gemm_kernel(const u16* __restrict__ A, const u16* __restrict__ W,
                const float* __restrict__ bias, float* __restrict__ out) {
  const int lane = threadIdx.x & 31, wave = threadIdx.x >> 5;
  const int mBase = blockIdx.x * 64 + (wave >> 2) * 32;
  const int nBase = blockIdx.y * 256 + (wave & 3) * 64;
  const int nl = lane & 15, hi16 = lane >> 4;
  v8f acc[2][4] = {};
  for (int k0 = 0; k0 < kE; k0 += 32) {
    v16bf a0 = asbf(ldA_bf(A + (size_t)mBase * kE + k0, kE, lane));
    v16bf a1 = asbf(ldA_bf(A + (size_t)(mBase + 16) * kE + k0, kE, lane));
    v16bf b[4];
#pragma unroll
    for (int ni = 0; ni < 4; ++ni)
      b[ni] = asbf(ldB_bf(W + (size_t)(nBase + ni * 16 + nl) * kE + k0, lane));
#pragma unroll
    for (int ni = 0; ni < 4; ++ni) {
      acc[0][ni] = wmma_bf16(a0, b[ni], acc[0][ni]);
      acc[1][ni] = wmma_bf16(a1, b[ni], acc[1][ni]);
    }
  }
#pragma unroll
  for (int mi = 0; mi < 2; ++mi)
#pragma unroll
    for (int ni = 0; ni < 4; ++ni)
#pragma unroll
      for (int j = 0; j < 8; ++j) {
        int m = mBase + mi * 16 + j + 8 * hi16;
        int n = nBase + ni * 16 + nl;
        out[(size_t)m * kE + n] = acc[mi][ni][j] + bias[n];
      }
}

extern "C" void kernel_launch(void* const* d_in, const int* in_sizes, int n_in,
                              void* d_out, int out_size, void* d_ws, size_t ws_size,
                              hipStream_t stream) {
  (void)in_sizes; (void)n_in; (void)out_size; (void)ws_size;
  const float* q  = (const float*)d_in[0];
  const float* k  = (const float*)d_in[1];
  const float* v  = (const float*)d_in[2];
  const float* Wq = (const float*)d_in[3];
  const float* bq = (const float*)d_in[4];
  const float* Wk = (const float*)d_in[5];
  const float* bk = (const float*)d_in[6];
  const float* Wv = (const float*)d_in[7];
  const float* bv = (const float*)d_in[8];
  const float* Wo = (const float*)d_in[9];
  const float* bo = (const float*)d_in[10];
  const float* rk = (const float*)d_in[11];
  const float* rv = (const float*)d_in[12];
  float* out = (float*)d_out;

  char* ws = (char*)d_ws;
  size_t off = 0;
  auto carve = [&](size_t bytes) -> char* {
    char* p = ws + off;
    off += (bytes + 255) & ~(size_t)255;
    return p;
  };
  u16* WqB = (u16*)carve((size_t)kE * kE * 2);
  u16* WkB = (u16*)carve((size_t)kE * kE * 2);
  u16* WvB = (u16*)carve((size_t)kE * kE * 2);
  u16* WoB = (u16*)carve((size_t)kE * kE * 2);
  u16* rkB = (u16*)carve((size_t)kR * kD * 2);
  u16* rvT = (u16*)carve((size_t)kD * kRp * 2);
  u16* xqB = (u16*)carve((size_t)kM * kE * 2);
  u16* xkB = (u16*)carve((size_t)kM * kE * 2);
  u16* xvB = (u16*)carve((size_t)kM * kE * 2);
  u16* qhB = (u16*)carve((size_t)kB * kH * kS * kD * 2);
  u16* khB = (u16*)carve((size_t)kB * kH * kS * kD * 2);
  u16* vhT = (u16*)carve((size_t)kB * kH * kS * kD * 2);
  u16* ctx = (u16*)carve((size_t)kM * kE * 2);

  const int nW = kE * kE;          // 1M
  const int nX = kM * kE;          // 4M
  cvt_bf16_kernel<<<(nW + 255) / 256, 256, 0, stream>>>(Wq, WqB, nW);
  cvt_bf16_kernel<<<(nW + 255) / 256, 256, 0, stream>>>(Wk, WkB, nW);
  cvt_bf16_kernel<<<(nW + 255) / 256, 256, 0, stream>>>(Wv, WvB, nW);
  cvt_bf16_kernel<<<(nW + 255) / 256, 256, 0, stream>>>(Wo, WoB, nW);
  cvt_bf16_kernel<<<(nX + 255) / 256, 256, 0, stream>>>(q, xqB, nX);
  cvt_bf16_kernel<<<(nX + 255) / 256, 256, 0, stream>>>(k, xkB, nX);
  cvt_bf16_kernel<<<(nX + 255) / 256, 256, 0, stream>>>(v, xvB, nX);
  cvt_bf16_kernel<<<(kR * kD + 255) / 256, 256, 0, stream>>>(rk, rkB, kR * kD);
  cvt_rvT_kernel<<<(kD * kRp + 255) / 256, 256, 0, stream>>>(rv, rvT);

  proj_kernel<<<dim3(kM / 64, kE / 256, 3), 256, 0, stream>>>(
      xqB, xkB, xvB, WqB, WkB, WvB, bq, bk, bv, qhB, khB, vhT);

  size_t smem = (size_t)16 * kS * 4 + (size_t)16 * kS * 2 + (size_t)16 * kD * 2 +
                (size_t)16 * kQp * 4 + (size_t)16 * kRp * 4 +
                (size_t)8 * 16 * 64 * 4 + (size_t)16 * kD * 4 + 64;  // 271424 B
  attn_kernel<<<dim3(kB * kH * (kS / 16)), 256, smem, stream>>>(
      qhB, khB, vhT, rkB, rvT, ctx);

  out_gemm_kernel<<<dim3(kM / 64, kE / 256), 256, 0, stream>>>(ctx, WoB, bo, out);
}
